// BahdanauAttention_58703613001785
// MI455X (gfx1250) — matrix operand order repaired and checked by reading, not compiled, not run on hardware
//
#include <hip/hip_runtime.h>
#include <hip/hip_bf16.h>
#include <math.h>

// ---------------------------------------------------------------------------
// Bahdanau attention, MI455X (gfx1250, wave32, WMMA).
// B=64, T=512, H=1024, UNITS=1024.
// Outputs: context [B,H] (65536 f32) then attn [B,T] (32768 f32), concatenated.
//
// Exact simplification: softmax over T cancels dec_proj (constant per batch
// over T) and bv, so attn/context depend only on tanh(enc@Wh + bh) @ Wv.
// ---------------------------------------------------------------------------

#define Bsz   64
#define Tlen  512
#define Hdim  1024
#define Udim  1024
#define NROWS (Bsz * Tlen)          // 32768 rows of the big GEMM
#define RPB   32                    // rows per block in score kernel

typedef __attribute__((ext_vector_type(16))) __bf16 v16bf;
typedef __attribute__((ext_vector_type(8)))  float  v8f;

// ------------------------------------------------------------------
// Kernel 1: Wh [H][U] f32 (row-major)  ->  WhT [U][H] bf16 (row-major)
// ------------------------------------------------------------------
__global__ void pack_wht(const float* __restrict__ Wh, __bf16* __restrict__ WhT) {
    __shared__ __bf16 tile[64][65];
    const int h0 = blockIdx.x * 64;
    const int u0 = blockIdx.y * 64;
    const int tid = threadIdx.x;
    for (int i = tid; i < 64 * 64; i += 256) {
        int r = i >> 6;       // h offset
        int c = i & 63;       // u offset
        tile[c][r] = (__bf16)Wh[(size_t)(h0 + r) * Udim + (u0 + c)];
    }
    __syncthreads();
    for (int i = tid; i < 64 * 64; i += 256) {
        int r = i >> 6;       // u offset
        int c = i & 63;       // h offset
        WhT[(size_t)(u0 + r) * Hdim + (h0 + c)] = tile[r][c];
    }
}

// ------------------------------------------------------------------
// Kernel 2: fused  score[row] = sum_u tanh((enc@Wh)[row,u] + bh[u]) * Wv[u]
// One block per 32-row tile; 8 waves; wave w owns U-tiles [8w, 8w+8).
// Each B fragment feeds TWO 16x16x32 WMMAs (row subtiles 0 and 1) -> halves
// WhT L2 traffic. Fragments loaded just-in-time (clean codegen, no spills);
// latency hidden by 8 waves/block across the WGP.
// ------------------------------------------------------------------
__global__ void __launch_bounds__(256)
score_wmma(const float* __restrict__ enc,     // [B*T, H] f32
           const __bf16* __restrict__ WhT,    // [U, H] bf16
           const float* __restrict__ bh,      // [U]
           const float* __restrict__ Wv,      // [U]
           float* __restrict__ scores)        // [B*T]
{
    __shared__ __bf16 sA[RPB][Hdim];          // 64 KB A tile, bf16
    __shared__ float  sScore[RPB];

    const int tid = threadIdx.x;
    const int R0  = blockIdx.x * RPB;         // first global row of this tile

    if (tid < RPB) sScore[tid] = 0.0f;

    // Stage + convert the 32x1024 A tile (coalesced f32 reads).
    for (int i = tid; i < RPB * Hdim; i += 256) {
        int r = i >> 10;
        int h = i & (Hdim - 1);
        sA[r][h] = (__bf16)enc[(size_t)(R0 + r) * Hdim + h];
    }
    __syncthreads();

    const int wave = tid >> 5;
    const int lane = tid & 31;
    const int m    = lane & 15;               // A row / B column within tile
    const int half = lane >> 4;               // 0: lanes 0-15, 1: lanes 16-31

    float rowAcc[2][8];
    #pragma unroll
    for (int s = 0; s < 2; ++s)
        #pragma unroll
        for (int j = 0; j < 8; ++j) rowAcc[s][j] = 0.0f;

    union Frag { unsigned int u[8]; v16bf v; };

    for (int utl = 0; utl < 8; ++utl) {
        const int n0 = (wave * 8 + utl) * 16; // U-tile base column
        const int n  = n0 + m;
        const __bf16* bpBase = WhT + (size_t)n * Hdim + 8 * half;

        v8f acc0 = {}, acc1 = {};

        #pragma unroll 4
        for (int k0 = 0; k0 < Hdim; k0 += 32) {
            // ---- B fragment (32x16 bf16): lane m holds column n;
            //      VGPR v: K = (v<4 ? 2v : 16+2(v-4)) + 8*half, packed pairs (contiguous in WhT).
            Frag fb;
            const __bf16* bp = bpBase + k0;
            #pragma unroll
            for (int v = 0; v < 8; ++v)
                fb.u[v] = *(const unsigned int*)(bp + ((v < 4) ? (2 * v) : (16 + 2 * (v - 4))));

            // ---- A fragments (16x32 bf16): lane m holds row m;
            //      VGPR v: K = (v<4 ? 2v : 16+2(v-4)) + 8*half, packed pairs.
            Frag fa0, fa1;
            #pragma unroll
            for (int v = 0; v < 8; ++v) {
                int kk = k0 + ((v < 4) ? (2 * v) : (16 + 2 * (v - 4))) + 8 * half;
                fa0.u[v] = *(const unsigned int*)&sA[m][kk];
                fa1.u[v] = *(const unsigned int*)&sA[16 + m][kk];
            }

            acc0 = __builtin_amdgcn_wmma_f32_16x16x32_bf16(
                false, fa0.v, false, fb.v, (short)0, acc0, false, false);
            acc1 = __builtin_amdgcn_wmma_f32_16x16x32_bf16(
                false, fa1.v, false, fb.v, (short)0, acc1, false, false);
        }

        // Epilogue: tanh + bias, dot with Wv slice, accumulate per row.
        const float wv  = Wv[n];
        const float bhv = bh[n];
        #pragma unroll
        for (int j = 0; j < 8; ++j) {
            // C layout: VGPR j -> row (j + 8*half) of the subtile, col n.
            rowAcc[0][j] += tanhf(acc0[j] + bhv) * wv;
            rowAcc[1][j] += tanhf(acc1[j] + bhv) * wv;
        }
    }

    // Reduce across the 16 lanes sharing each row (xor 1,2,4,8 stays in half).
    #pragma unroll
    for (int s = 0; s < 2; ++s) {
        #pragma unroll
        for (int j = 0; j < 8; ++j) {
            float v = rowAcc[s][j];
            v += __shfl_xor(v, 1, 32);
            v += __shfl_xor(v, 2, 32);
            v += __shfl_xor(v, 4, 32);
            v += __shfl_xor(v, 8, 32);
            if (m == 0) atomicAdd(&sScore[s * 16 + j + 8 * half], v); // ds_add_f32
        }
    }
    __syncthreads();
    if (tid < RPB) scores[R0 + tid] = sScore[tid];
}

// ------------------------------------------------------------------
// Kernel 3: softmax over T per batch. 64 blocks x 256 threads (2 elems/thread).
// ------------------------------------------------------------------
__global__ void __launch_bounds__(256)
softmax_t(const float* __restrict__ scores, float* __restrict__ attn) {
    __shared__ float red[256];
    const int b   = blockIdx.x;
    const int tid = threadIdx.x;
    const float s0 = scores[b * Tlen + tid];
    const float s1 = scores[b * Tlen + 256 + tid];

    red[tid] = fmaxf(s0, s1);
    __syncthreads();
    for (int off = 128; off > 0; off >>= 1) {
        if (tid < off) red[tid] = fmaxf(red[tid], red[tid + off]);
        __syncthreads();
    }
    const float mx = red[0];
    __syncthreads();

    const float e0 = __expf(s0 - mx);
    const float e1 = __expf(s1 - mx);
    red[tid] = e0 + e1;
    __syncthreads();
    for (int off = 128; off > 0; off >>= 1) {
        if (tid < off) red[tid] += red[tid + off];
        __syncthreads();
    }
    const float inv = 1.0f / red[0];

    attn[b * Tlen + tid]       = e0 * inv;
    attn[b * Tlen + 256 + tid] = e1 * inv;
}

// ------------------------------------------------------------------
// Kernel 4: context[b,h] = sum_t attn[b,t] * enc[b,t,h].  Bandwidth-bound.
// Grid (64), 256 threads, float4 per thread (coalesced b128 along h).
// ------------------------------------------------------------------
__global__ void __launch_bounds__(256)
context_sum(const float* __restrict__ enc, const float* __restrict__ attn,
            float* __restrict__ ctx) {
    const int b  = blockIdx.x;
    const int h4 = threadIdx.x;               // 256 * 4 = 1024 h's
    const float4* ep = (const float4*)(enc + (size_t)b * Tlen * Hdim) + h4;
    const float*  ap = attn + b * Tlen;
    float4 acc = make_float4(0.f, 0.f, 0.f, 0.f);
    #pragma unroll 4
    for (int t = 0; t < Tlen; ++t) {
        const float  a = ap[t];
        const float4 e = ep[(size_t)t * (Hdim / 4)];
        acc.x += a * e.x; acc.y += a * e.y;
        acc.z += a * e.z; acc.w += a * e.w;
    }
    ((float4*)(ctx + b * Hdim))[h4] = acc;
}

// ---------------------------------------------------------------------------
extern "C" void kernel_launch(void* const* d_in, const int* in_sizes, int n_in,
                              void* d_out, int out_size, void* d_ws, size_t ws_size,
                              hipStream_t stream) {
    (void)in_sizes; (void)n_in; (void)out_size; (void)ws_size;
    // inputs: 0=dec_hidden, 1=enc_output, 2=Wh, 3=bh, 4=Ws, 5=bs, 6=Wv, 7=bv
    const float* enc = (const float*)d_in[1];
    const float* Wh  = (const float*)d_in[2];
    const float* bh  = (const float*)d_in[3];
    const float* Wv  = (const float*)d_in[6];
    // dec_hidden / Ws / bs / bv cancel exactly in the T-softmax -> unused.

    float* ctx  = (float*)d_out;                       // [B,H]
    float* attn = (float*)d_out + Bsz * Hdim;          // [B,T]

    __bf16* WhT   = (__bf16*)d_ws;                                   // 2 MB
    float* scores = (float*)((char*)d_ws + (size_t)Udim * Hdim * 2); // 128 KB

    pack_wht   <<<dim3(16, 16), 256, 0, stream>>>(Wh, WhT);
    score_wmma <<<NROWS / RPB,  256, 0, stream>>>(enc, WhT, bh, Wv, scores);
    softmax_t  <<<Bsz,          256, 0, stream>>>(scores, attn);
    context_sum<<<Bsz,          256, 0, stream>>>(enc, attn, ctx);
}
